// Net_4922032521431
// MI455X (gfx1250) — matrix operand order repaired and checked by reading, not compiled, not run on hardware
//
#include <hip/hip_runtime.h>
#include <hip/hip_bf16.h>

// ---------------------------------------------------------------------------
// Problem constants (from reference)
// ---------------------------------------------------------------------------
#define BB      256      // batch
#define NN      784      // nodes
#define EE      6272     // edges
#define CC      32       // GraphConv channels
#define HH      512      // fc1 hidden
#define NOUT    10
#define KTOT    (NN*CC)  // 25088 = fc1 reduction dim

typedef __attribute__((ext_vector_type(8)))  float          v8f;
typedef __attribute__((ext_vector_type(16))) __bf16         v16bf;
typedef __attribute__((ext_vector_type(8)))  unsigned short v8us;
typedef __attribute__((ext_vector_type(16))) unsigned short v16us;
typedef int vi4 __attribute__((vector_size(4 * sizeof(int))));

#define AS1 __attribute__((address_space(1)))
#define AS3 __attribute__((address_space(3)))

// ---- CDNA5 async global->LDS DMA (ASYNCcnt-tracked), with safe fallback ----
#if defined(__has_builtin)
#  if __has_builtin(__builtin_amdgcn_global_load_async_to_lds_b128)
#    define HAVE_ASYNC_LDS 1
#  endif
#  if __has_builtin(__builtin_amdgcn_s_wait_asynccnt)
#    define HAVE_WAIT_ASYNC 1
#  endif
#endif
#ifndef HAVE_ASYNC_LDS
#  define HAVE_ASYNC_LDS 0
#endif
#ifndef HAVE_WAIT_ASYNC
#  define HAVE_WAIT_ASYNC 0
#endif

__device__ __forceinline__ void async_copy16(const void* gsrc, void* ldst) {
#if HAVE_ASYNC_LDS
    __builtin_amdgcn_global_load_async_to_lds_b128(
        (AS1 vi4*)(vi4*)gsrc, (AS3 vi4*)(vi4*)ldst, 0, 0);
#else
    *(uint4*)ldst = *(const uint4*)gsrc;
#endif
}
__device__ __forceinline__ void wait_async() {
#if HAVE_WAIT_ASYNC
    __builtin_amdgcn_s_wait_asynccnt(0);
#elif HAVE_ASYNC_LDS
    asm volatile("s_wait_asynccnt 0" ::: "memory");
#endif
}

// fp32 -> bf16 bits, round-to-nearest-even
__device__ __forceinline__ unsigned short f2bf(float f) {
    union { float f; unsigned u; } v; v.f = f;
    unsigned r = v.u + 0x7FFFu + ((v.u >> 16) & 1u);
    return (unsigned short)(r >> 16);
}
__device__ __forceinline__ float bf2f(unsigned short u) {
    union { unsigned u; float f; } v; v.u = ((unsigned)u) << 16;
    return v.f;
}
__device__ __forceinline__ float elu(float x) {
    return x > 0.f ? x : (__expf(x) - 1.f);
}

__device__ __forceinline__ v8f wmma_bf16(v16us a, v16us b, v8f c) {
    return __builtin_amdgcn_wmma_f32_16x16x32_bf16(
        /*neg_a=*/false, __builtin_bit_cast(v16bf, a),
        /*neg_b=*/false, __builtin_bit_cast(v16bf, b),
        /*c_mod=*/(short)0, c, /*reuse_a=*/false, /*reuse_b=*/false);
}

// ---------------------------------------------------------------------------
// Kernel 1: GraphConv1.  One block per batch element.
//   agg[n] = sum_e val[e]*x[b, col[e]]  (F==1, so SpMM is scalar per node)
//   h1[b,n,c] = elu(agg[n]*W1[c] + b1[c])  -> bf16
// ---------------------------------------------------------------------------
__global__ void gconv1_kernel(const float* __restrict__ x,
                              const int*   __restrict__ erow,
                              const int*   __restrict__ ecol,
                              const float* __restrict__ eval,
                              const float* __restrict__ W1,
                              const float* __restrict__ b1,
                              unsigned short* __restrict__ h1) {
    __shared__ float xl[NN];
    __shared__ float agg[NN];
    __shared__ float w1l[CC];
    __shared__ float b1l[CC];
    const int b = blockIdx.x, tid = threadIdx.x;

    for (int i = tid; i < NN; i += 256) {
        xl[i]  = x[(size_t)b * NN + i];
        agg[i] = 0.f;
    }
    if (tid < CC) { w1l[tid] = W1[tid]; b1l[tid] = b1[tid]; }
    __syncthreads();

    for (int e = tid; e < EE; e += 256)
        atomicAdd(&agg[erow[e]], eval[e] * xl[ecol[e]]);
    __syncthreads();

    for (int i = tid; i < NN * CC; i += 256) {
        int n = i >> 5, c = i & 31;
        h1[(size_t)b * KTOT + i] = f2bf(elu(agg[n] * w1l[c] + b1l[c]));
    }
}

// ---------------------------------------------------------------------------
// Kernel 2: GraphConv2.  One block (256 thr / 8 waves) per batch element.
//   Dynamic LDS (~149KB of the 320KB WGP LDS):
//     hin  : h1[b]  bf16   784*32        (50176 B)   <- async DMA from HBM
//     agg  : fp32 accumulator 784*32     (100352 B)
//     w2l  : W2 bf16, WMMA-B layout [n][k] (2048 B)
//     b2l  : bias (128 B)
//   SpMM: wave-per-edge, lane = channel, ds_add_f32 atomics.
//   GEMM: 49 M-tiles x 2 N-tiles of v_wmma_f32_16x16x32_bf16, then ELU,
//   store h2 bf16 in flattened [b, n*32+c] layout for fc1.
// ---------------------------------------------------------------------------
__global__ void gconv2_kernel(const unsigned short* __restrict__ h1,
                              const int*   __restrict__ erow,
                              const int*   __restrict__ ecol,
                              const float* __restrict__ eval,
                              const float* __restrict__ W2,
                              const float* __restrict__ b2,
                              unsigned short* __restrict__ h2) {
    extern __shared__ __align__(128) char smem[];
    unsigned short* hin = (unsigned short*)smem;                  // 50176 B
    float*          agg = (float*)(smem + 50176);                 // 100352 B
    unsigned short* w2l = (unsigned short*)(smem + 150528);       // 2048 B
    float*          b2l = (float*)(smem + 152576);                // 128 B

    const int b = blockIdx.x, tid = threadIdx.x;
    const int wid = tid >> 5, lane = tid & 31;

    // stage h1[b] into LDS via async DMA (overlaps with agg zero / W2 prep)
    {
        const unsigned short* src = h1 + (size_t)b * KTOT;
        for (int i = tid; i < KTOT / 8; i += 256)
            async_copy16(src + i * 8, hin + i * 8);
    }
    for (int i = tid; i < KTOT; i += 256) agg[i] = 0.f;
    // W2 -> bf16, transposed into WMMA-B layout: w2l[n*32 + k'] = W2[k][n], k' = k with bits 3,4 swapped
    for (int i = tid; i < CC * CC; i += 256) {
        int n = i >> 5, k = i & 31;
        w2l[n * 32 + ((k & 7) | ((k & 8) << 1) | ((k & 16) >> 1))] = f2bf(W2[k * CC + n]);
    }
    if (tid < CC) b2l[tid] = b2[tid];
    wait_async();
    __syncthreads();

    // SpMM: one wave per edge, lane = channel (ds_add_f32, conflict-free banks)
    for (int e = wid; e < EE; e += 8) {
        int r = erow[e], c = ecol[e];
        float v = eval[e];
        atomicAdd(&agg[r * CC + lane], v * bf2f(hin[c * CC + lane]));
    }
    __syncthreads();

    // GEMM [784,32] @ [32,32] : 49*2 = 98 WMMA tiles over 8 waves
    const int hi = lane >> 4, lo = lane & 15;
    for (int t = wid; t < 49 * 2; t += 8) {
        int mt = t >> 1, nt = t & 1;
        // B fragment: lane -> col lo (+nt*16), K = 8*hi+[0,8) then 16+8*hi+[0,8)
        v16us bf = *(const v16us*)&w2l[(nt * 16 + lo) * 32 + hi * 16];
        // A fragment: row = mt*16+lo, K halves [kb,kb+8) and [kb+16,kb+24)
        v16us af;
        const float* pa = &agg[(mt * 16 + lo) * 32 + hi * 8];
        #pragma unroll
        for (int j = 0; j < 8; ++j) {
            af[j]     = f2bf(pa[j]);
            af[8 + j] = f2bf(pa[16 + j]);
        }
        v8f acc = {};
        acc = wmma_bf16(af, bf, acc);
        // D layout: m = r + 8*hi, n = lo   -> elu + bias -> bf16 store
        #pragma unroll
        for (int r = 0; r < 8; ++r) {
            int m = mt * 16 + hi * 8 + r;   // node index
            int n = nt * 16 + lo;           // channel
            h2[(size_t)b * KTOT + m * CC + n] = f2bf(elu(acc[r] + b2l[n]));
        }
    }
}

// ---------------------------------------------------------------------------
// Kernel 3a: zero the fc1 accumulator (split-K partials land here)
// ---------------------------------------------------------------------------
__global__ void zero_kernel(float* __restrict__ z, int n) {
    int i = blockIdx.x * blockDim.x + threadIdx.x;
    if (i < n) z[i] = 0.f;
}

// ---------------------------------------------------------------------------
// Kernel 3b: fc1  z[256,512] += flat[256,25088] @ Wf1[25088,512]
//   grid = (32 N-stripes of 16 cols) x (8 K-chunks of 3136).
//   Block (256 thr / 8 waves) covers all 256 batch rows of its stripe;
//   wave w owns rows [32w, 32w+32) as two 16x16 WMMA accumulators.
//   Double-buffered software pipeline:
//     A tile (256x32 bf16, 16KB) : async global->LDS DMA (ASYNCcnt)
//     B tile (Wf1 fp32 -> bf16 transposed into WMMA-B layout, 1KB) : VALU
//   Issue next tiles, run WMMAs on current, s_wait_asynccnt + one barrier.
//   Wf1 is read exactly once from HBM (the 51MB / ~2.2us roofline floor).
// ---------------------------------------------------------------------------
__global__ void fc1_kernel(const unsigned short* __restrict__ h2,
                           const float* __restrict__ Wf1,
                           float* __restrict__ z) {
    __shared__ __align__(32) unsigned short Alds[2][256 * 32]; // 2 x 16 KB
    __shared__ __align__(32) unsigned short Blds[2][16 * 32];  // 2 x 1 KB

    const int n0   = blockIdx.x * 16;
    const int kbeg = blockIdx.y * (KTOT / 8);
    const int kend = kbeg + (KTOT / 8);
    const int tid  = threadIdx.x;
    const int wid  = tid >> 5, lane = tid & 31;
    const int hi = lane >> 4, lo = lane & 15;
    const int m0 = wid * 32;

    v8f acc0 = {}, acc1 = {};

    // stage A tile: 256 rows x 32 k bf16 = 1024 x 16B chunks (async DMA)
    auto stageA = [&](int k0, int buf) {
        #pragma unroll
        for (int c = 0; c < 4; ++c) {
            int ch  = tid + c * 256;
            int row = ch >> 2, part = ch & 3;
            async_copy16(&h2[(size_t)row * KTOT + k0 + part * 8],
                         &Alds[buf][row * 32 + part * 8]);
        }
    };
    // stage B tile: Wf1[k0..k0+32, n0..n0+16] fp32 -> bf16, WMMA-B layout
    auto stageB = [&](int k0, int buf) {
        #pragma unroll
        for (int c = 0; c < 2; ++c) {
            int e  = tid + c * 256;
            int kk = e >> 4, nn = e & 15;
            Blds[buf][nn * 32 + ((kk & 7) | ((kk & 8) << 1) | ((kk & 16) >> 1))] = f2bf(Wf1[(size_t)(k0 + kk) * HH + n0 + nn]);
        }
    };

    stageA(kbeg, 0);
    stageB(kbeg, 0);
    wait_async();
    __syncthreads();

    int buf = 0;
    for (int k0 = kbeg; k0 < kend; k0 += 32) {
        // kick off next tile while the matrix pipe chews on this one
        if (k0 + 32 < kend) {
            stageA(k0 + 32, buf ^ 1);
            stageB(k0 + 32, buf ^ 1);
        }

        // ---- fragments + WMMA on current buffer ----
        v16us bfrag = *(const v16us*)&Blds[buf][lo * 32 + hi * 16];
        {
            const unsigned short* pa = &Alds[buf][(m0 + lo) * 32 + hi * 8];
            v8us a0 = *(const v8us*)pa;
            v8us a1 = *(const v8us*)(pa + 16);
            v16us af = __builtin_shufflevector(a0, a1, 0,1,2,3,4,5,6,7,
                                                      8,9,10,11,12,13,14,15);
            acc0 = wmma_bf16(af, bfrag, acc0);
        }
        {
            const unsigned short* pa = &Alds[buf][(m0 + 16 + lo) * 32 + hi * 8];
            v8us a0 = *(const v8us*)pa;
            v8us a1 = *(const v8us*)(pa + 16);
            v16us af = __builtin_shufflevector(a0, a1, 0,1,2,3,4,5,6,7,
                                                      8,9,10,11,12,13,14,15);
            acc1 = wmma_bf16(af, bfrag, acc1);
        }

        wait_async();
        __syncthreads();
        buf ^= 1;
    }

    // ---- flush split-K partials: global_atomic_add_f32 ----
    const int n = n0 + lo;
    #pragma unroll
    for (int r = 0; r < 8; ++r) {
        int m = m0 + hi * 8 + r;
        atomicAdd(&z[(size_t)m * HH + n],        acc0[r]);
        atomicAdd(&z[(size_t)(m + 16) * HH + n], acc1[r]);
    }
}

// ---------------------------------------------------------------------------
// Kernel 4: head.  relu(z + bf1) @ Wf2 + bf2, softmax.  One wave per row.
// ---------------------------------------------------------------------------
__global__ void head_kernel(const float* __restrict__ z,
                            const float* __restrict__ bf1,
                            const float* __restrict__ Wf2,
                            const float* __restrict__ bf2,
                            float* __restrict__ out) {
    __shared__ float zl[HH];
    __shared__ float logits[NOUT];
    const int b = blockIdx.x, lane = threadIdx.x;

    for (int i = lane; i < HH; i += 32)
        zl[i] = fmaxf(0.f, z[(size_t)b * HH + i] + bf1[i]);  // fused bias+relu
    __syncthreads();

    if (lane < NOUT) {
        float s = bf2[lane];
        for (int k = 0; k < HH; ++k) s += zl[k] * Wf2[k * NOUT + lane];
        logits[lane] = s;
    }
    __syncthreads();

    if (lane == 0) {
        float mx = logits[0];
        #pragma unroll
        for (int i = 1; i < NOUT; ++i) mx = fmaxf(mx, logits[i]);
        float e[NOUT], sum = 0.f;
        #pragma unroll
        for (int i = 0; i < NOUT; ++i) { e[i] = __expf(logits[i] - mx); sum += e[i]; }
        float inv = 1.f / sum;
        #pragma unroll
        for (int i = 0; i < NOUT; ++i) out[(size_t)b * NOUT + i] = e[i] * inv;
    }
}

// ---------------------------------------------------------------------------
// Launch
// ---------------------------------------------------------------------------
extern "C" void kernel_launch(void* const* d_in, const int* in_sizes, int n_in,
                              void* d_out, int out_size, void* d_ws, size_t ws_size,
                              hipStream_t stream) {
    const float* x    = (const float*)d_in[0];
    const int*   erow = (const int*)  d_in[1];
    const int*   ecol = (const int*)  d_in[2];
    const float* eval = (const float*)d_in[3];
    const float* W1   = (const float*)d_in[4];
    const float* b1   = (const float*)d_in[5];
    const float* W2   = (const float*)d_in[6];
    const float* b2   = (const float*)d_in[7];
    const float* Wf1  = (const float*)d_in[8];
    const float* bf1  = (const float*)d_in[9];
    const float* Wf2  = (const float*)d_in[10];
    const float* bf2v = (const float*)d_in[11];

    char* ws = (char*)d_ws;
    const size_t h_bytes = (size_t)BB * KTOT * sizeof(unsigned short); // 12.8 MB
    unsigned short* h1 = (unsigned short*)ws;
    unsigned short* h2 = (unsigned short*)(ws + h_bytes);
    float*          zb = (float*)(ws + 2 * h_bytes);                   // 256*512 fp32

    gconv1_kernel<<<BB, 256, 0, stream>>>(x, erow, ecol, eval, W1, b1, h1);
    gconv2_kernel<<<BB, 256, 152704, stream>>>(h1, erow, ecol, eval, W2, b2, h2);
    zero_kernel<<<(BB * HH + 255) / 256, 256, 0, stream>>>(zb, BB * HH);
    fc1_kernel<<<dim3(HH / 16, 8), 256, 0, stream>>>(h2, Wf1, zb);
    head_kernel<<<BB, 32, 0, stream>>>(zb, bf1, Wf2, bf2v, (float*)d_out);
}
